// EGNNDecoder_88502096101687
// MI455X (gfx1250) — matrix operand order repaired and checked by reading, not compiled, not run on hardware
//
#include <hip/hip_runtime.h>

// ---------------------------------------------------------------------------
// EGNN decoder for MI455X (gfx1250, wave32, WMMA).
// B=64 molecules x NA=64 atoms, H=128, L=6 layers, dense intra-molecule graph.
// One workgroup (8 wave32) per molecule; all activations + hot edge weights
// live in LDS across all 6 layers. GEMMs use v_wmma_f32_16x16x32_bf16.
// Weights are pre-swizzled into B-fragment order so every WMMA operand is
// loaded with two 128-bit loads (no per-element repacking). gemm is fully
// templated so tile epilogues have no runtime branches.
// ---------------------------------------------------------------------------

typedef __bf16 bf16;
typedef __attribute__((ext_vector_type(16))) __bf16 v16bf;
typedef __attribute__((ext_vector_type(8)))  float  v8f;
typedef __attribute__((ext_vector_type(4)))  unsigned int u32x4;

#define TPB 256            // 8 waves of 32
#define NWAVE 8

__device__ __forceinline__ bf16 f2bf(float f) {
  union { float f; unsigned u; } in; in.f = f;
  unsigned r = in.u + 0x7FFFu + ((in.u >> 16) & 1u);   // round-to-nearest-even
  union { unsigned short s; bf16 b; } out;
  out.s = (unsigned short)(r >> 16);
  return out.b;
}
__device__ __forceinline__ float bf2f(bf16 x) {
  union { bf16 b; unsigned short s; } in; in.b = x;
  union { unsigned u; float f; } out;
  out.u = ((unsigned)in.s) << 16;
  return out.f;
}
// silu via v_exp_f32 + v_rcp_f32 (no IEEE divide sequence in the epilogue)
__device__ __forceinline__ float silu_f(float x) {
  return x * __builtin_amdgcn_rcpf(1.0f + __expf(-x));
}

// --- WMMA fragment loaders ---------------------------------------------------
// A 16x32 (MxK) from row-major LDS: lane slots are two contiguous 16B runs.
__device__ __forceinline__ v16bf load_a_bf(const bf16* A, int lda,
                                           int mbase, int kbase, int lane) {
  const bf16* row = A + (mbase + (lane & 15)) * lda + kbase + ((lane >> 4) << 3);
  union { u32x4 q[2]; v16bf v; } u;
  u.q[0] = *(const u32x4*)(row);
  u.q[1] = *(const u32x4*)(row + 16);
  return u.v;
}
// B 32x16 (KxN) pre-swizzled: tile (kt,nt) stored as 32 lanes x 16 bf16,
// lane-major (32B contiguous per lane) -> two b128 loads.
__device__ __forceinline__ v16bf load_b_packed(const bf16* Bp, int ntiles,
                                               int kb, int nb, int lane) {
  const bf16* blob = Bp + (((size_t)((kb >> 5) * ntiles + (nb >> 4))) * 32 + lane) * 16;
  union { u32x4 q[2]; v16bf v; } u;
  u.q[0] = *(const u32x4*)(blob);
  u.q[1] = *(const u32x4*)(blob + 8);
  return u.v;
}

// D = A(MxK) * Bp(KxN, packed) + bias, optional SiLU; f32 and/or bf16 out.
// Template params fold every epilogue branch at compile time.
template <bool HASB, bool ACT, bool OUTF, bool OUTB>
__device__ __forceinline__ void gemm(const bf16* __restrict__ A, int lda,
                                     const bf16* __restrict__ Bp,
                                     int M, int N, int K,
                                     const float* __restrict__ bias,
                                     float* __restrict__ Cf, bf16* __restrict__ Cb,
                                     int ldc) {
  const int lane = threadIdx.x & 31;
  const int wave = threadIdx.x >> 5;
  const int nt = N >> 4;
  const int tiles = (M >> 4) * nt;
  for (int t = wave; t < tiles; t += NWAVE) {
    const int mb = (t / nt) << 4;
    const int nb = (t % nt) << 4;
    v8f acc = {0.f, 0.f, 0.f, 0.f, 0.f, 0.f, 0.f, 0.f};
    for (int kb = 0; kb < K; kb += 32) {
      v16bf a = load_a_bf(A, lda, mb, kb, lane);
      v16bf b = load_b_packed(Bp, nt, kb, nb, lane);
      acc = __builtin_amdgcn_wmma_f32_16x16x32_bf16(
          false, a, false, b, (short)0, acc, false, false);
    }
    const int n = nb + (lane & 15);
    const int mr = mb + ((lane >> 4) << 3);
    const float bs = HASB ? bias[n] : 0.0f;
#pragma unroll
    for (int r = 0; r < 8; ++r) {
      float v = acc[r] + bs;
      if (ACT) v = silu_f(v);
      const int m = mr + r;
      if (OUTF) Cf[m * ldc + n] = v;
      if (OUTB) Cb[m * ldc + n] = f2bf(v);
    }
  }
}

// cooperative LDS staging copy (b128 in / b128 out)
__device__ __forceinline__ void copy_lds(bf16* __restrict__ dst,
                                         const bf16* __restrict__ src, int nelem) {
  u32x4* d = (u32x4*)dst;
  const u32x4* s = (const u32x4*)src;
  for (int i = threadIdx.x; i < (nelem >> 3); i += TPB) d[i] = s[i];
}

// --- weight f32 -> bf16 pack into B-fragment order ---------------------------
// dst layout: layer-major, tiles kt-major then nt, each tile = 32 lanes x 16.
__global__ void pack_b_kernel(const float* __restrict__ s, bf16* __restrict__ d,
                              int K, int Kpad, int N, int layers, int lstride) {
  long i = (long)blockIdx.x * blockDim.x + threadIdx.x;
  const int ntiles = N >> 4;
  const long tilesPerLayer = (long)(Kpad >> 5) * ntiles;
  const long total = (long)layers * tilesPerLayer * 512;
  if (i >= total) return;
  int slot = (int)(i & 15);
  int lane = (int)((i >> 4) & 31);
  long tile = i >> 9;
  int l  = (int)(tile / tilesPerLayer);
  long tl = tile % tilesPerLayer;
  int kt = (int)(tl / ntiles), nt = (int)(tl % ntiles);
  int k = (kt << 5) + ((lane >> 4) << 3) + (slot & 7) + ((slot >> 3) << 4);
  int n = (nt << 4) + (lane & 15);
  float v = (k < K) ? s[(size_t)l * lstride + (size_t)k * N + n] : 0.0f;
  d[i] = f2bf(v);
}

// --- kernel parameter block --------------------------------------------------
struct P {
  const float *z, *atom_types, *coords_init;
  const float *li_b1, *li_b2, *li_b3;
  const float *eb1, *eb2, *eb3, *nb1, *nb2, *nb3, *cb1, *cw2, *lng, *lnb;
  const float *ew1f;   // f32 edge_w1 (row 256 = d^2 column)
  const bf16 *li_w1p, *li_w2p, *li_w3p;
  const bf16 *ew1p, *ew2p, *ew3p, *nw1p, *nw2p, *nw3p, *cw1p;
  float *h0;           // [N,128] workspace
  float *out;          // [B*NA*3]
};

// --- kernel 1: h0 = MLP3([atom_types | z]) ----------------------------------
__global__ __launch_bounds__(TPB) void latent_mlp(P p) {
  extern __shared__ char smem[];
  bf16* X  = (bf16*)smem;        // [64][160]  (144 real cols, K-padded)
  bf16* a1 = X + 64 * 160;       // [64][256]
  bf16* a2 = a1 + 64 * 256;      // [64][128]
  const int tid = threadIdx.x, mol = blockIdx.x;
  for (int idx = tid; idx < 64 * 160; idx += TPB) {
    int a = idx / 160, c = idx % 160;
    float v = 0.0f;
    if (c < 16)       v = p.atom_types[(mol * 64 + a) * 16 + c];
    else if (c < 144) v = p.z[mol * 128 + (c - 16)];
    X[idx] = f2bf(v);
  }
  __syncthreads();
  gemm<true, true, false, true>(X, 160, p.li_w1p, 64, 256, 160, p.li_b1,
                                nullptr, a1, 256);
  __syncthreads();
  gemm<true, true, false, true>(a1, 256, p.li_w2p, 64, 128, 256, p.li_b2,
                                nullptr, a2, 128);
  __syncthreads();
  gemm<true, false, true, false>(a2, 128, p.li_w3p, 64, 128, 128, p.li_b3,
                                 p.h0 + mol * 64 * 128, nullptr, 128);
}

// --- kernel 2: all 6 EGNN layers, one molecule per workgroup ----------------
__global__ __launch_bounds__(TPB) void egnn_layers(P p) {
  extern __shared__ char smem[];
  bf16*  hbf  = (bf16*)smem;          // [64][128] h (bf16, GEMM input)
  bf16*  HW   = hbf  + 64 * 128;      // [64][256] h @ ew1[128:256]   (j term)
  bf16*  TI   = HW   + 64 * 256;      // [64][256] h @ ew1[0:128]+eb1 (i term)
  bf16*  act1 = TI   + 64 * 256;      // [32][256] edge hidden1 tile
  bf16*  act2 = act1 + 32 * 256;      // [32][128] edge hidden2 / coord hidden
  bf16*  msgb = act2 + 32 * 128;      // [32][128] msg tile
  bf16*  ew2s = msgb + 32 * 128;      // [256x128] staged packed ew2 (64 KB)
  bf16*  ew3s = ew2s + 256 * 128;     // [128x128] staged packed ew3 (32 KB)
  bf16*  cw1s = ew3s + 128 * 128;     // [128x128] staged packed cw1 (32 KB)
  float* agg  = (float*)(cw1s + 128 * 128); // [64][128] message sums / hu
  float* hf   = agg  + 64 * 128;      // [64][128] h (f32, residual+LN)
  float* w1c  = hf   + 64 * 128;      // [256] d^2 weight row (f32)
  float* crd  = w1c  + 256;           // [64][4]
  float* cacc = crd  + 64 * 4;        // [64][4]
  float* cwb  = cacc + 64 * 4;        // [32]

  const int tid = threadIdx.x, mol = blockIdx.x;

  for (int idx = tid; idx < 64 * 128; idx += TPB) {
    float v = p.h0[mol * 64 * 128 + idx];
    hf[idx] = v; hbf[idx] = f2bf(v);
  }
  if (tid < 64) {
    crd[tid * 4 + 0] = p.coords_init[(mol * 64 + tid) * 3 + 0];
    crd[tid * 4 + 1] = p.coords_init[(mol * 64 + tid) * 3 + 1];
    crd[tid * 4 + 2] = p.coords_init[(mol * 64 + tid) * 3 + 2];
    crd[tid * 4 + 3] = 0.0f;
  }
  __syncthreads();

  for (int l = 0; l < 6; ++l) {
    const bf16* ew1 = p.ew1p + l * 256 * 256;   // packed; first half = rows 0..127
    const bf16* nw1 = p.nw1p + l * 256 * 256;
    const bf16* nw2 = p.nw2p + l * 256 * 128;
    const bf16* nw3 = p.nw3p + l * 128 * 128;
    const float* eb1 = p.eb1 + l * 256;
    const float* eb2 = p.eb2 + l * 128;
    const float* eb3 = p.eb3 + l * 128;
    const float* nb1 = p.nb1 + l * 256;
    const float* nb2 = p.nb2 + l * 128;
    const float* nb3 = p.nb3 + l * 128;
    const float* cb1 = p.cb1 + l * 128;
    const float* cw2 = p.cw2 + l * 128;
    const float* lng = p.lng + l * 128;
    const float* lnb = p.lnb + l * 128;
    const float* w1l = p.ew1f + l * 257 * 256 + 256 * 256; // d^2 row of ew1 (f32)

    // Split first edge-MLP layer: ef@W1 = TI[i] + HW[j] + d2*w1l (once/layer).
    gemm<true,  false, false, true>(hbf, 128, ew1, 64, 256, 128, eb1,
                                    nullptr, TI, 256);
    gemm<false, false, false, true>(hbf, 128, ew1 + 128 * 256, 64, 256, 128,
                                    nullptr, nullptr, HW, 256);
    // Stage hot edge-loop weights into LDS (read 128x per layer per WG).
    copy_lds(ew2s, p.ew2p + l * 256 * 128, 256 * 128);
    copy_lds(ew3s, p.ew3p + l * 128 * 128, 128 * 128);
    copy_lds(cw1s, p.cw1p + l * 128 * 128, 128 * 128);
    for (int idx = tid; idx < 64 * 128; idx += TPB) agg[idx] = 0.0f;
    w1c[tid] = w1l[tid];            // 256 threads == 256 entries
    if (tid < 64) { cacc[tid*4] = 0.f; cacc[tid*4+1] = 0.f; cacc[tid*4+2] = 0.f; }
    __syncthreads();

    for (int i = 0; i < 64; ++i) {
      const float cix = crd[i*4], ciy = crd[i*4+1], ciz = crd[i*4+2];
      const int n = tid;                       // each thread owns one column
      const float ti = bf2f(TI[i * 256 + n]);
      const float wl = w1c[n];
      for (int half = 0; half < 2; ++half) {
        const int jb = half * 32;
        // act1[j'][n] = silu(HW[j] + TI[i] + d2(i,j)*w1l[n])
        for (int jj = 0; jj < 32; ++jj) {
          const int j = jb + jj;
          float dx = cix - crd[j*4], dy = ciy - crd[j*4+1], dz = ciz - crd[j*4+2];
          float d2 = dx*dx + dy*dy + dz*dz;
          float v = bf2f(HW[j * 256 + n]) + ti + d2 * wl;
          act1[jj * 256 + n] = f2bf(silu_f(v));
        }
        __syncthreads();
        gemm<true, true, false, true>(act1, 256, ew2s, 32, 128, 256, eb2,
                                      nullptr, act2, 128);
        __syncthreads();
        gemm<true, false, false, true>(act2, 128, ew3s, 32, 128, 128, eb3,
                                       nullptr, msgb, 128);
        __syncthreads();
        // agg[i] += sum_{j != i} msg[j]
        if (tid < 128) {
          float s = agg[i * 128 + tid];
          for (int jj = 0; jj < 32; ++jj)
            if (jb + jj != i) s += bf2f(msgb[jj * 128 + tid]);
          agg[i * 128 + tid] = s;
        }
        __syncthreads();
        // coord weights: cw = silu(msg @ cw1 + cb1) @ cw2
        gemm<true, true, false, true>(msgb, 128, cw1s, 32, 128, 128, cb1,
                                      nullptr, act2, 128);
        __syncthreads();
        if (tid < 32) {
          float s = 0.0f;
          for (int k = 0; k < 128; ++k) s += bf2f(act2[tid * 128 + k]) * cw2[k];
          cwb[tid] = s;
        }
        __syncthreads();
        if (tid < 3) {
          float s = 0.0f;
          for (int jj = 0; jj < 32; ++jj) {
            int j = jb + jj;
            if (j == i) continue;
            s += cwb[jj] * (crd[i*4 + tid] - crd[j*4 + tid]);
          }
          cacc[i*4 + tid] += s;
        }
        __syncthreads();
      }
    }

    // node MLP: nx = [h | agg]  (reuse TI as nx, HW as hidden1, act1 as hidden2)
    for (int idx = tid; idx < 64 * 128; idx += TPB) {
      int a = idx >> 7, c = idx & 127;
      TI[a * 256 + c]       = hbf[idx];
      TI[a * 256 + 128 + c] = f2bf(agg[idx]);
    }
    __syncthreads();
    gemm<true, true, false, true>(TI, 256, nw1, 64, 256, 256, nb1,
                                  nullptr, HW, 256);
    __syncthreads();
    gemm<true, true, false, true>(HW, 256, nw2, 64, 128, 256, nb2,
                                  nullptr, act1, 128);
    __syncthreads();
    gemm<true, false, true, false>(act1, 128, nw3, 64, 128, 128, nb3,
                                   agg, nullptr, 128); // hu -> agg
    __syncthreads();

    // residual + LayerNorm; coords += cacc
    if (tid < 64) {
      const int a = tid;
      float sum = 0.0f, sq = 0.0f;
      for (int c = 0; c < 128; ++c) {
        float x = hf[a * 128 + c] + agg[a * 128 + c];
        sum += x; sq += x * x;
      }
      float mean = sum * (1.0f / 128.0f);
      float var  = sq * (1.0f / 128.0f) - mean * mean;
      float rs = rsqrtf(var + 1e-5f);
      for (int c = 0; c < 128; ++c) {
        float x = hf[a * 128 + c] + agg[a * 128 + c];
        float y = (x - mean) * rs * lng[c] + lnb[c];
        hf[a * 128 + c] = y; hbf[a * 128 + c] = f2bf(y);
      }
    } else if (tid < 128) {
      const int a = tid - 64;
      crd[a*4+0] += cacc[a*4+0];
      crd[a*4+1] += cacc[a*4+1];
      crd[a*4+2] += cacc[a*4+2];
    }
    __syncthreads();
  }

  if (tid < 64) {
    p.out[(mol * 64 + tid) * 3 + 0] = crd[tid * 4 + 0];
    p.out[(mol * 64 + tid) * 3 + 1] = crd[tid * 4 + 1];
    p.out[(mol * 64 + tid) * 3 + 2] = crd[tid * 4 + 2];
  }
}

// ---------------------------------------------------------------------------
extern "C" void kernel_launch(void* const* d_in, const int* in_sizes, int n_in,
                              void* d_out, int out_size, void* d_ws, size_t ws_size,
                              hipStream_t stream) {
  (void)in_sizes; (void)n_in; (void)out_size; (void)ws_size;
  const float* z      = (const float*)d_in[1];
  const float* atoms  = (const float*)d_in[2];
  const float* coords = (const float*)d_in[3];
  const float* li_w1  = (const float*)d_in[4];
  const float* li_b1  = (const float*)d_in[5];
  const float* li_w2  = (const float*)d_in[6];
  const float* li_b2  = (const float*)d_in[7];
  const float* li_w3  = (const float*)d_in[8];
  const float* li_b3  = (const float*)d_in[9];
  const float* ew1 = (const float*)d_in[10]; const float* eb1 = (const float*)d_in[11];
  const float* ew2 = (const float*)d_in[12]; const float* eb2 = (const float*)d_in[13];
  const float* ew3 = (const float*)d_in[14]; const float* eb3 = (const float*)d_in[15];
  const float* nw1 = (const float*)d_in[16]; const float* nb1 = (const float*)d_in[17];
  const float* nw2 = (const float*)d_in[18]; const float* nb2 = (const float*)d_in[19];
  const float* nw3 = (const float*)d_in[20]; const float* nb3 = (const float*)d_in[21];
  const float* cw1 = (const float*)d_in[22]; const float* cb1 = (const float*)d_in[23];
  const float* cw2 = (const float*)d_in[24];
  const float* lng = (const float*)d_in[25]; const float* lnb = (const float*)d_in[26];

  char* ws = (char*)d_ws;
  size_t off = 0;
  auto carve = [&](size_t bytes) {
    void* q = ws + off;
    off = (off + bytes + 255) & ~(size_t)255;
    return q;
  };
  bf16* li_w1p = (bf16*)carve(160 * 256 * 2);
  bf16* li_w2p = (bf16*)carve(256 * 128 * 2);
  bf16* li_w3p = (bf16*)carve(128 * 128 * 2);
  bf16* ew1p   = (bf16*)carve((size_t)6 * 256 * 256 * 2);  // rows 0..255 only
  bf16* ew2p   = (bf16*)carve((size_t)6 * 256 * 128 * 2);
  bf16* ew3p   = (bf16*)carve((size_t)6 * 128 * 128 * 2);
  bf16* nw1p   = (bf16*)carve((size_t)6 * 256 * 256 * 2);
  bf16* nw2p   = (bf16*)carve((size_t)6 * 256 * 128 * 2);
  bf16* nw3p   = (bf16*)carve((size_t)6 * 128 * 128 * 2);
  bf16* cw1p   = (bf16*)carve((size_t)6 * 128 * 128 * 2);
  float* h0    = (float*)carve((size_t)4096 * 128 * 4);

  auto pack = [&](const float* s, bf16* d, int K, int Kpad, int N, int layers,
                  int lstride) {
    long total = (long)layers * Kpad * N;
    pack_b_kernel<<<(int)((total + 255) / 256), 256, 0, stream>>>(
        s, d, K, Kpad, N, layers, lstride);
  };
  pack(li_w1, li_w1p, 144, 160, 256, 1, 144 * 256);
  pack(li_w2, li_w2p, 256, 256, 128, 1, 256 * 128);
  pack(li_w3, li_w3p, 128, 128, 128, 1, 128 * 128);
  pack(ew1, ew1p, 256, 256, 256, 6, 257 * 256);   // skip d^2 row (handled in f32)
  pack(ew2, ew2p, 256, 256, 128, 6, 256 * 128);
  pack(ew3, ew3p, 128, 128, 128, 6, 128 * 128);
  pack(nw1, nw1p, 256, 256, 256, 6, 256 * 256);
  pack(nw2, nw2p, 256, 256, 128, 6, 256 * 128);
  pack(nw3, nw3p, 128, 128, 128, 6, 128 * 128);
  pack(cw1, cw1p, 128, 128, 128, 6, 128 * 128);

  P p;
  p.z = z; p.atom_types = atoms; p.coords_init = coords;
  p.li_b1 = li_b1; p.li_b2 = li_b2; p.li_b3 = li_b3;
  p.eb1 = eb1; p.eb2 = eb2; p.eb3 = eb3;
  p.nb1 = nb1; p.nb2 = nb2; p.nb3 = nb3;
  p.cb1 = cb1; p.cw2 = cw2; p.lng = lng; p.lnb = lnb;
  p.ew1f = ew1;
  p.li_w1p = li_w1p; p.li_w2p = li_w2p; p.li_w3p = li_w3p;
  p.ew1p = ew1p; p.ew2p = ew2p; p.ew3p = ew3p;
  p.nw1p = nw1p; p.nw2p = nw2p; p.nw3p = nw3p; p.cw1p = cw1p;
  p.h0 = h0; p.out = (float*)d_out;

  // Dynamic LDS: latent = 68 KB; layers ~= 307 KB (ISA: <=320 KB per WG)
  size_t lds_latent = (size_t)(64*160 + 64*256 + 64*128) * 2;
  size_t lds_layers =
      (size_t)(64*128 + 2*64*256 + 32*256 + 2*32*128      // activations (bf16)
               + 256*128 + 2*128*128) * 2                 // staged weights (bf16)
    + (size_t)(2*64*128 + 256 + 2*64*4 + 32) * 4;         // f32 buffers
  latent_mlp<<<64, TPB, lds_latent, stream>>>(p);
  egnn_layers<<<64, TPB, lds_layers, stream>>>(p);
}
